// MinGRU_72086731096321
// MI455X (gfx1250) — compile-verified, matrix operand order checked by reading, not run on hardware
//
#include <hip/hip_runtime.h>

// MinGRU forward for MI455X (gfx1250, wave32).
//   Phase 0: fp32 -> bf16 hi/lo split pre-pass (bandwidth bound).
//   Phase 1: hg = x @ W^T (M=32768, N=2048, K=1024) on v_wmma_f32_16x16x32_bf16
//            with bf16x3 split precision, fused log-space epilogue.
//   Phases 2-4: chunked logaddexp scan (h_t = a_t h_{t-1} + b_t in log space).

typedef __attribute__((ext_vector_type(16))) __bf16 v16bf;
typedef __attribute__((ext_vector_type(8)))  float  v8f;

namespace {
constexpr int kB = 8;
constexpr int kS = 4096;
constexpr int kD = 1024;
constexpr int kM = kB * kS;          // 32768 flattened rows
constexpr int kChunks = 64;          // chunks along S
constexpr int kClen = kS / kChunks;  // 64 steps per chunk
constexpr float kNegInf = -1.0e30f;  // finite "-inf" sentinel (no NaN paths)
}

__device__ __forceinline__ float log_add_exp(float u, float v) {
  float mx = fmaxf(u, v);
  float mn = fminf(u, v);
  return mx + log1pf(__expf(mn - mx));
}

// ---------------------------------------------------------------------------
// Phase 0: split fp32 array into bf16 hi + bf16 lo arrays (8 elems/thread).
// ---------------------------------------------------------------------------
__global__ __launch_bounds__(256)
void split_bf16_kernel(const float* __restrict__ src, __bf16* __restrict__ hi,
                       __bf16* __restrict__ lo) {
  const size_t i = ((size_t)blockIdx.x * blockDim.x + threadIdx.x) * 8;
  float4 f0 = *reinterpret_cast<const float4*>(src + i);
  float4 f1 = *reinterpret_cast<const float4*>(src + i + 4);
  union { __bf16 h[8]; uint4 q; } uh, ul;
  float f[8] = {f0.x, f0.y, f0.z, f0.w, f1.x, f1.y, f1.z, f1.w};
#pragma unroll
  for (int j = 0; j < 8; ++j) {
    __bf16 h_ = (__bf16)f[j];
    uh.h[j] = h_;
    ul.h[j] = (__bf16)(f[j] - (float)h_);
  }
  *reinterpret_cast<uint4*>(hi + i) = uh.q;
  *reinterpret_cast<uint4*>(lo + i) = ul.q;
}

// ---------------------------------------------------------------------------
// Phase 1: GEMM + fused log-space epilogue.
// Fragment layout per CDNA5 ISA 7.12.2 (16-bit A 16x32; B mirrors it because
// W is [N,K] row-major == column-major B): lane L holds row (L&15), K chunks
// [koff, koff+8) and [koff+16, koff+24), koff = (L<16) ? 0 : 8.
// ---------------------------------------------------------------------------
union FragU { uint4 q[2]; v16bf v; };

__device__ __forceinline__ v16bf load_frag(const __bf16* __restrict__ row, int kbase) {
  FragU u;
  u.q[0] = *reinterpret_cast<const uint4*>(row + kbase);        // K chunk 0 (8 bf16)
  u.q[1] = *reinterpret_cast<const uint4*>(row + kbase + 16);   // K chunk 1 (8 bf16)
  return u.v;
}

__device__ __forceinline__ void emit_logspace(float h, float g, int m, int d,
                                              float* lc_buf, float* lv_buf) {
  // softplus(g) = max(g,0) + log1p(exp(-|g|))
  float sp0   = log1pf(__expf(-fabsf(g)));
  float lc    = -(fmaxf(g, 0.0f) + sp0);       // log(1 - z) = -softplus(g)
  float log_z = fminf(g, 0.0f) - sp0;          // log(z)     = -softplus(-g)
  // log_g(h): h>=0 -> log(h + 0.5) ; h<0 -> -softplus(-h) = h - log1p(exp(h))
  float lth = (h >= 0.0f) ? __logf(h + 0.5f) : (h - log1pf(__expf(h)));
  size_t idx = (size_t)m * kD + d;
  lc_buf[idx] = lc;
  lv_buf[idx] = log_z + lth;
}

#define WMMA_BF16(A, Bm, C) \
  __builtin_amdgcn_wmma_f32_16x16x32_bf16(false, (A), false, (Bm), (short)0, (C), false, false)

// Workgroup = 256 threads = 8 waves; WG tile = 64(M) x 64(d), hidden+gate
// columns paired so the epilogue has both operands in registers.
__global__ __launch_bounds__(256)
void mingru_gemm_epilogue(const __bf16* __restrict__ xh, const __bf16* __restrict__ xl,
                          const __bf16* __restrict__ wh_, const __bf16* __restrict__ wl_,
                          float* __restrict__ lc_buf, float* lv_buf) {
  const int tid  = threadIdx.x;
  const int lane = tid & 31;
  const int wave = tid >> 5;
  const int wm   = wave >> 2;                   // 0..1  (M halves)
  const int wn   = wave & 3;                    // 0..3  (N quarters)

  const int m_wave = blockIdx.x * 64 + wm * 32; // first row of this wave's 2 subtiles
  const int d_col  = blockIdx.y * 64 + wn * 16; // first hidden column

  const int lrow = lane & 15;
  const int koff = (lane < 16) ? 0 : 8;

  const __bf16* xr0h = xh + (size_t)(m_wave + lrow) * kD + koff;
  const __bf16* xr0l = xl + (size_t)(m_wave + lrow) * kD + koff;
  const __bf16* xr1h = xh + (size_t)(m_wave + 16 + lrow) * kD + koff;
  const __bf16* xr1l = xl + (size_t)(m_wave + 16 + lrow) * kD + koff;
  const __bf16* whh  = wh_ + (size_t)(d_col + lrow) * kD + koff;        // hidden rows
  const __bf16* whl  = wl_ + (size_t)(d_col + lrow) * kD + koff;
  const __bf16* wgh  = wh_ + (size_t)(d_col + kD + lrow) * kD + koff;   // gate rows
  const __bf16* wgl  = wl_ + (size_t)(d_col + kD + lrow) * kD + koff;

  v8f acc_h0 = {}, acc_h1 = {}, acc_g0 = {}, acc_g1 = {};

  for (int k0 = 0; k0 < kD; k0 += 32) {
    v16bf a0h = load_frag(xr0h, k0);
    v16bf a0l = load_frag(xr0l, k0);
    v16bf a1h = load_frag(xr1h, k0);
    v16bf a1l = load_frag(xr1l, k0);
    v16bf bhh = load_frag(whh, k0);
    v16bf bhl = load_frag(whl, k0);
    v16bf bgh = load_frag(wgh, k0);
    v16bf bgl = load_frag(wgl, k0);

    // Split precision: hi*hi + hi*lo + lo*hi (lo*lo ~ 2^-32, dropped).
    acc_h0 = WMMA_BF16(a0h, bhh, acc_h0);
    acc_h1 = WMMA_BF16(a1h, bhh, acc_h1);
    acc_g0 = WMMA_BF16(a0h, bgh, acc_g0);
    acc_g1 = WMMA_BF16(a1h, bgh, acc_g1);

    acc_h0 = WMMA_BF16(a0h, bhl, acc_h0);
    acc_h1 = WMMA_BF16(a1h, bhl, acc_h1);
    acc_g0 = WMMA_BF16(a0h, bgl, acc_g0);
    acc_g1 = WMMA_BF16(a1h, bgl, acc_g1);

    acc_h0 = WMMA_BF16(a0l, bhh, acc_h0);
    acc_h1 = WMMA_BF16(a1l, bhh, acc_h1);
    acc_g0 = WMMA_BF16(a0l, bgh, acc_g0);
    acc_g1 = WMMA_BF16(a1l, bgh, acc_g1);
  }

  // C/D layout (ISA 7.12.2): lane L -> column (L&15); VGPR v -> row v (+8 for
  // lanes 16..31). Fused epilogue: log_coeffs -> lc_buf, log_values -> lv_buf.
  const int mofs = (lane < 16) ? 0 : 8;
  const int dcol = d_col + lrow;
#pragma unroll
  for (int v = 0; v < 8; ++v) {
    int m0 = m_wave + v + mofs;
    int m1 = m_wave + 16 + v + mofs;
    emit_logspace(acc_h0[v], acc_g0[v], m0, dcol, lc_buf, lv_buf);
    emit_logspace(acc_h1[v], acc_g1[v], m1, dcol, lc_buf, lv_buf);
  }
}

// ---------------------------------------------------------------------------
// Phase 2: per-(b, chunk, d) compose the 64-step segment into (A, LB):
//   logh_out = logaddexp(A + logh_in, LB).
// ---------------------------------------------------------------------------
__global__ __launch_bounds__(256)
void mingru_chunk_reduce(const float* __restrict__ lc, const float* __restrict__ lv,
                         float* __restrict__ carryA, float* __restrict__ carryB) {
  const int d = blockIdx.x * blockDim.x + threadIdx.x; // 0..1023
  const int c = blockIdx.y;                            // 0..63
  const int b = blockIdx.z;                            // 0..7
  size_t base = ((size_t)b * kS + (size_t)c * kClen) * kD + d;
  float A = 0.0f, LB = kNegInf;
  for (int t = 0; t < kClen; ++t) {
    float lcv = lc[base + (size_t)t * kD];
    float lvv = lv[base + (size_t)t * kD];
    LB = log_add_exp(LB + lcv, lvv);
    A += lcv;
  }
  size_t ci = ((size_t)b * kChunks + c) * kD + d;
  carryA[ci] = A;
  carryB[ci] = LB;
}

// Phase 3: scan the 64 chunk carries per (b,d); store the carry-in of each chunk.
__global__ __launch_bounds__(256)
void mingru_carry_scan(const float* __restrict__ carryA, const float* __restrict__ carryB,
                       float* __restrict__ hin) {
  const int idx = blockIdx.x * blockDim.x + threadIdx.x; // 0..8191
  const int b = idx / kD;
  const int d = idx % kD;
  float logh = kNegInf;
  for (int c = 0; c < kChunks; ++c) {
    size_t ci = ((size_t)b * kChunks + c) * kD + d;
    hin[ci] = logh;
    logh = log_add_exp(carryA[ci] + logh, carryB[ci]);
  }
}

// Phase 4: rescan each chunk with its carry-in; write exp(log_h) in place
// over log_values (lv_out aliases d_out; each element is read then written by
// the same thread in the same iteration, so aliasing is safe).
__global__ __launch_bounds__(256)
void mingru_apply(const float* __restrict__ lc, float* lv_out,
                  const float* __restrict__ hin) {
  const int d = blockIdx.x * blockDim.x + threadIdx.x;
  const int c = blockIdx.y;
  const int b = blockIdx.z;
  size_t base = ((size_t)b * kS + (size_t)c * kClen) * kD + d;
  float logh = hin[((size_t)b * kChunks + c) * kD + d];
  for (int t = 0; t < kClen; ++t) {
    size_t i = base + (size_t)t * kD;
    float lcv = lc[i];
    float lvv = lv_out[i];
    logh = log_add_exp(logh + lcv, lvv);
    lv_out[i] = __expf(logh);
  }
}

extern "C" void kernel_launch(void* const* d_in, const int* in_sizes, int n_in,
                              void* d_out, int out_size, void* d_ws, size_t ws_size,
                              hipStream_t stream) {
  (void)in_sizes; (void)n_in; (void)out_size; (void)ws_size;
  const float* x = (const float*)d_in[0];   // [8, 4096, 1024] fp32
  const float* w = (const float*)d_in[1];   // [2048, 1024] fp32
  float* out = (float*)d_out;               // [8, 4096, 1024] fp32

  // Workspace layout:
  //   lc     : kM*kD floats            (134 MB)
  //   x_hi   : kM*kD bf16              ( 64 MB)
  //   x_lo   : kM*kD bf16              ( 64 MB)
  //   w_hi   : 2*kD*kD bf16            (  4 MB)
  //   w_lo   : 2*kD*kD bf16            (  4 MB)
  //   carryA / carryB / hin : kB*kChunks*kD floats each (2 MB each)
  float*  lc   = (float*)d_ws;
  __bf16* x_hi = (__bf16*)(lc + (size_t)kM * kD);
  __bf16* x_lo = x_hi + (size_t)kM * kD;
  __bf16* w_hi = x_lo + (size_t)kM * kD;
  __bf16* w_lo = w_hi + (size_t)2 * kD * kD;
  float*  carryA = (float*)(w_lo + (size_t)2 * kD * kD);
  float*  carryB = carryA + (size_t)kB * kChunks * kD;
  float*  hin    = carryB + (size_t)kB * kChunks * kD;

  // Phase 0: fp32 -> bf16 hi/lo split (x: 33.5M elems, W: 2M elems).
  split_bf16_kernel<<<(size_t)kM * kD / (256 * 8), 256, 0, stream>>>(x, x_hi, x_lo);
  split_bf16_kernel<<<(size_t)2 * kD * kD / (256 * 8), 256, 0, stream>>>(w, w_hi, w_lo);

  // Phase 1: WMMA GEMM + log-space epilogue. log_values lives in d_out.
  mingru_gemm_epilogue<<<dim3(kM / 64, kD / 64), 256, 0, stream>>>(
      x_hi, x_lo, w_hi, w_lo, lc, out);
  // Phase 2: per-chunk segment composition.
  mingru_chunk_reduce<<<dim3(kD / 256, kChunks, kB), 256, 0, stream>>>(lc, out, carryA, carryB);
  // Phase 3: scan chunk carries (8192 sequences x 64 steps).
  mingru_carry_scan<<<(kB * kD) / 256, 256, 0, stream>>>(carryA, carryB, hin);
  // Phase 4: apply carries, write h = exp(log_h) in place.
  mingru_apply<<<dim3(kD / 256, kChunks, kB), 256, 0, stream>>>(lc, out, hin);
}
